// ExpMovAvgModel_26980984553960
// MI455X (gfx1250) — compile-verified, matrix operand order checked by reading, not run on hardware
//
#include <hip/hip_runtime.h>

typedef _Float16 v16h __attribute__((ext_vector_type(16)));
typedef _Float16 v8h  __attribute__((ext_vector_type(8)));
typedef _Float16 v4h  __attribute__((ext_vector_type(4)));
typedef float    v8f  __attribute__((ext_vector_type(8)));
typedef float    v4f  __attribute__((ext_vector_type(4)));

#define NB 32
#define NT 1024
#define NH 128

// workspace layout (bytes)
#define XH_BYTES  (NB * NT * NH * 2)            // 8 MB of f16 gathered embeddings
#define INV_OFF   (XH_BYTES)                    // B*T f32 inverse magnitudes
#define LAM_OFF   (XH_BYTES + NB * NT * 4)      // B*T f32 lambda

static __device__ __forceinline__ v16h combine16(v8h lo, v8h hi) {
    v16h r;
#pragma unroll
    for (int i = 0; i < 8; ++i) { r[i] = lo[i]; r[i + 8] = hi[i]; }
    return r;
}

// Inclusive suffix sum within each 16-lane DPP row: lane n -> sum of lanes n..15.
// row_shl:d (ctrl 0x100|d) pulls from lane n+d, bound_ctrl=1 zero-fills past row end.
static __device__ __forceinline__ float suffix_add16(float v) {
    int x;
    x = __builtin_amdgcn_update_dpp(0, __float_as_int(v), 0x101, 0xf, 0xf, true);
    v += __int_as_float(x);
    x = __builtin_amdgcn_update_dpp(0, __float_as_int(v), 0x102, 0xf, 0xf, true);
    v += __int_as_float(x);
    x = __builtin_amdgcn_update_dpp(0, __float_as_int(v), 0x104, 0xf, 0xf, true);
    v += __int_as_float(x);
    x = __builtin_amdgcn_update_dpp(0, __float_as_int(v), 0x108, 0xf, 0xf, true);
    v += __int_as_float(x);
    return v;
}

// Broadcast lane 0 of each 16-lane row to all lanes of that row (one VALU op).
static __device__ __forceinline__ float bcast_row0(float v) {
    const int vi = __float_as_int(v);
    return __int_as_float(
        __builtin_amdgcn_permlane16(vi, vi, 0u, 0u, false, false));
}

// ---------------------------------------------------------------------------
// Kernel 1: gather + precompute. One wave (32 lanes) per token row.
//   xh[row]     = (f16) embd[seq[row]]
//   invmag[row] = 1/||embd[seq[row]]||
//   lam[row]    = exp(embd[seq[row]] . lam_w + lam_b)
// ---------------------------------------------------------------------------
__global__ __launch_bounds__(256)
void gather_precompute_kernel(const int* __restrict__ seq,
                              const float* __restrict__ ew,
                              const float* __restrict__ lam_w,
                              const float* __restrict__ lam_b,
                              _Float16* __restrict__ xh,
                              float* __restrict__ invmag,
                              float* __restrict__ lam) {
    const int wid  = threadIdx.x >> 5;
    const int lane = threadIdx.x & 31;
    const int row  = blockIdx.x * 8 + wid;              // < NB*NT
    const int p    = seq[row];

    v4f e = *(const v4f*)(ew + (size_t)p * NH + lane * 4);
    v4f w = *(const v4f*)(lam_w + lane * 4);

    float ss = e[0]*e[0] + e[1]*e[1] + e[2]*e[2] + e[3]*e[3];
    float dd = e[0]*w[0] + e[1]*w[1] + e[2]*w[2] + e[3]*w[3];
#pragma unroll
    for (int m = 16; m >= 1; m >>= 1) {
        ss += __shfl_xor(ss, m, 32);
        dd += __shfl_xor(dd, m, 32);
    }

    v4h h;
#pragma unroll
    for (int i = 0; i < 4; ++i) h[i] = (_Float16)e[i];
    *(v4h*)(xh + (size_t)row * NH + lane * 4) = h;      // 8-byte store

    if (lane == 0) {
        invmag[row] = rsqrtf(ss);
        lam[row]    = __expf(dd + lam_b[0]);
    }
}

// ---------------------------------------------------------------------------
// Kernel 2: WMMA sim tiles + DPP suffix-cumsum decay attention.
// grid = (NT/128, NB); 256 threads = 8 waves; wave w owns query tile tt.
// Double-buffered LDS staging of 16x128 f16 key tiles.
// ---------------------------------------------------------------------------
__global__ __launch_bounds__(256)
void ema_attn_kernel(const float* __restrict__ y,
                     const _Float16* __restrict__ xh,
                     const float* __restrict__ invmag,
                     const float* __restrict__ lam,
                     float* __restrict__ out) {
    __shared__ v8h ldsB[2][16 * 16];                    // 2 x (16 rows x 128 f16) = 8 KB

    const int b    = blockIdx.y;
    const int tid  = threadIdx.x;
    const int wid  = tid >> 5;
    const int lane = tid & 31;
    const int half = lane >> 4;
    const int n    = lane & 15;                          // key column within tile
    const int tt   = blockIdx.x * 8 + wid;               // query tile index [0,64)
    const int t0   = tt * 16;

    const _Float16* xbase = xh + (size_t)b * NT * NH;

    // A fragments (query tile rows, K=128), resident for the whole loop.
    const v8h* arow = (const v8h*)(xbase + (size_t)(t0 + n) * NH);
    v16h afrag[4];
#pragma unroll
    for (int c = 0; c < 4; ++c)
        afrag[c] = combine16(arow[4 * c + half], arow[4 * c + 2 + half]);

    float invm_t[8], lam_t[8];
#pragma unroll
    for (int r = 0; r < 8; ++r) {
        const int trow = t0 + r + 8 * half;
        invm_t[r] = invmag[b * NT + trow];
        lam_t[r]  = lam[b * NT + trow];
    }

    float carry[8] = {}, wsum[8] = {}, ysum[8] = {};

    const int stmax = blockIdx.x * 8 + 7;                // highest s-tile this block needs
    const int rr = tid >> 4, cc = tid & 15;              // cooperative stage coords

    // prefetch first key tile into buffer 0 (coalesced b128 per thread)
    v8h stage = *(const v8h*)(xbase + (size_t)(stmax * 16 + rr) * NH + cc * 8);
    ldsB[0][rr * 16 + cc] = stage;
    int buf = 0;

    for (int st = stmax; st >= 0; --st) {
        if (st > 0)  // issue next tile's global load early; hides under compute
            stage = *(const v8h*)(xbase + (size_t)((st - 1) * 16 + rr) * NH + cc * 8);

        __syncthreads();                                 // ldsB[buf] ready for all waves

        if (st <= tt) {                                  // wave-uniform diagonal skip
            // scalar loads first: latency hides under fragment loads + WMMA
            const int   s      = st * 16 + n;
            const float invm_s = invmag[b * NT + s];
            const float ys     = y[b * NT + s];

            const v8h* Bt = ldsB[buf];
            v8f acc = {};
#pragma unroll
            for (int c = 0; c < 4; ++c) {
                v16h bfr = combine16(Bt[n * 16 + 4 * c + half],
                                     Bt[n * 16 + 4 * c + 2 + half]);
                acc = __builtin_amdgcn_wmma_f32_16x16x32_f16(
                    false, afrag[c], false, bfr, (short)0, acc, false, false);
            }

            const float invs2 = 0.5f * invm_s;           // fold the (.. +1)*0.5
#pragma unroll
            for (int r = 0; r < 8; ++r) {
                const int trow = t0 + r + 8 * half;
                float sim = fmaf(acc[r], invm_t[r] * invs2, 0.5f);
                if (s >= trow) sim = 0.0f;               // only bites on diag tile

                // reversed (suffix) cumsum over the 16 key lanes, pure VALU DPP
                const float sc    = suffix_add16(sim);
                const float total = bcast_row0(sc);      // tile row sum -> all lanes
                const float delta = carry[r] + sc;       // + already-processed s' > tile
                const float wgt   = sim * __expf(-lam_t[r] * delta);
                wsum[r]  += wgt;
                ysum[r]  += wgt * ys;
                carry[r] += total;
            }
        }

        if (st > 0)                                      // stage next tile into other buf
            ldsB[buf ^ 1][rr * 16 + cc] = stage;
        buf ^= 1;
    }

    // cross-lane reduction of per-column partials (lane 0 of each row gets total)
#pragma unroll
    for (int r = 0; r < 8; ++r) {
        const float ws = suffix_add16(wsum[r]);
        const float yv = suffix_add16(ysum[r]);
        if (n == 0) {
            float v = yv / (ws + 1e-6f);
            v = fminf(fmaxf(v, 0.01f), 0.99f);
            out[b * NT + t0 + r + 8 * half] = v;
        }
    }
}

// ---------------------------------------------------------------------------
extern "C" void kernel_launch(void* const* d_in, const int* in_sizes, int n_in,
                              void* d_out, int out_size, void* d_ws, size_t ws_size,
                              hipStream_t stream) {
    const float* y       = (const float*)d_in[0];        // [B,T]
    const int*   seq     = (const int*)d_in[1];          // [B,T]
    const float* ew      = (const float*)d_in[2];        // [N_PROBLEMS,128]
    const float* lam_w   = (const float*)d_in[3];        // [128]
    const float* lam_b   = (const float*)d_in[4];        // [1]
    float*       out     = (float*)d_out;                // [B,T]

    _Float16* xh     = (_Float16*)d_ws;
    float*    invmag = (float*)((char*)d_ws + INV_OFF);
    float*    lamv   = (float*)((char*)d_ws + LAM_OFF);

    // Phase 1: gather + norms + lambda. 32768 rows, 8 waves/block.
    gather_precompute_kernel<<<dim3((NB * NT) / 8), dim3(256), 0, stream>>>(
        seq, ew, lam_w, lam_b, xh, invmag, lamv);

    // Phase 2: tiled WMMA attention. 8 query-blocks per batch.
    ema_attn_kernel<<<dim3(NT / 128, NB), dim3(256), 0, stream>>>(
        y, xh, invmag, lamv, out);
}
